// Fixed_Partitions_8203387535714
// MI455X (gfx1250) — matrix operand order repaired and checked by reading, not compiled, not run on hardware
//
#include <hip/hip_runtime.h>
#include <hip/hip_bf16.h>
#include <stdint.h>

typedef __bf16 bf16;
typedef __attribute__((ext_vector_type(8)))  bf16  v8bf;
typedef __attribute__((ext_vector_type(16))) bf16  v16bf;
typedef __attribute__((ext_vector_type(8)))  float v8f;

#define WMMA_BF16(a, b, c) \
  __builtin_amdgcn_wmma_f32_16x16x32_bf16(false, (a), false, (b), (short)0, (c), false, false)

#if defined(__has_builtin)
#if __has_builtin(__builtin_amdgcn_global_load_async_to_lds_b128)
#define HAVE_ASYNC_LDS 1
#endif
#endif

// Builtin signature (from hipcc diagnostic): param0 = int4 addrspace(1)*,
// (global source), param1 = LDS destination, then imm offset + cpol.
typedef int v4i_t __attribute__((vector_size(4 * sizeof(int))));
typedef v4i_t __attribute__((address_space(1)))* gptr_v4i;
typedef v4i_t __attribute__((address_space(3)))* lptr_v4i;
#define AS_GLOBAL(p) ((gptr_v4i)(p))
#define AS_LOCAL(p)  ((lptr_v4i)(p))

static __device__ __forceinline__ void wait_async0() {
#if defined(__has_builtin) && __has_builtin(__builtin_amdgcn_s_wait_asynccnt)
  __builtin_amdgcn_s_wait_asynccnt(0);
#else
  asm volatile("s_wait_asynccnt 0x0" ::: "memory");
#endif
}

static __device__ __forceinline__ v16bf cat16(v8bf lo, v8bf hi) {
  return __builtin_shufflevector(lo, hi, 0,1,2,3,4,5,6,7,8,9,10,11,12,13,14,15);
}
// A fragment (16x32 bf16): per lane 16 contiguous K elements of one row
static __device__ __forceinline__ v16bf ldA(const bf16* p) {
  return cat16(*(const v8bf*)p, *(const v8bf*)(p + 8));
}
// B fragment (32x16 bf16): per lane 16 contiguous K elements of one column
static __device__ __forceinline__ v16bf ldB(const bf16* p) {
  return cat16(*(const v8bf*)p, *(const v8bf*)(p + 8));
}

// ---------------------------------------------------------------------------
// Kernel 1: pack weights (fp32 -> bf16) into per-lane B-fragment layout.
// Tile (kt,nt) covers K rows [kt*32, kt*32+32), N cols [nt*16, nt*16+16).
// Lane L holds column n = nt*16+(L&15), k-run kbase = kt*32+(L>>4)*16 .. +15.
// wqkv: 32*96 = 3072 tiles; wout: 16*64 = 1024 tiles (ids 3072..4095).
// ---------------------------------------------------------------------------
__global__ __launch_bounds__(256) void pack_weights_k(const float* __restrict__ wqkv,
                                                      const float* __restrict__ wout,
                                                      bf16* __restrict__ pack) {
  int wv = threadIdx.x >> 5, lane = threadIdx.x & 31;
  int tid = blockIdx.x * 8 + wv;                       // 0..4095
  const float* src; int ldn, kt, nt;
  if (tid < 3072) { kt = tid / 96; nt = tid % 96; src = wqkv; ldn = 1536; }
  else { int t2 = tid - 3072; kt = t2 / 64; nt = t2 % 64; src = wout; ldn = 1024; }
  int n = nt * 16 + (lane & 15);
  int kbase = kt * 32 + ((lane >> 4) << 4);
  v8bf lo, hi;
#pragma unroll
  for (int i = 0; i < 8; ++i) {
    lo[i] = (bf16)src[(size_t)(kbase + i) * ldn + n];
    hi[i] = (bf16)src[(size_t)(kbase + 8 + i) * ldn + n];
  }
  bf16* dst = pack + (size_t)tid * 512 + lane * 16;
  *(v8bf*)dst = lo;
  *(v8bf*)(dst + 8) = hi;
}

// ---------------------------------------------------------------------------
// Kernel 2: gather/patchify x -> tokens bf16 [512 windows][64 tokens][1024 dims]
// token t = ph*8+pw, dim d = p1*256 + p2*64 + c. One block per (window, ph):
// coalesced fp32 reads of 4 image rows, LDS transpose, coalesced bf16 writes.
// ---------------------------------------------------------------------------
__global__ __launch_bounds__(256) void gather_tokens_k(const float* __restrict__ x,
                                                       bf16* __restrict__ tok) {
  __shared__ float lds[64 * 132];                      // [c][p1*32 + w], pad to 132
  int blk = blockIdx.x;                                // win*8 + ph
  int win = blk >> 3, ph = blk & 7;
  int b = win >> 6, h1 = (win >> 3) & 7, w1 = win & 7;
  int tid = threadIdx.x;
#pragma unroll
  for (int i = 0; i < 32; ++i) {
    int idx = i * 256 + tid;                           // 8192 elements
    int c = idx >> 7, p1 = (idx >> 5) & 3, w = idx & 31;
    float v = x[(((size_t)(b * 64 + c)) * 256 + (h1 * 32 + ph * 4 + p1)) * 256
                + (w1 * 32 + w)];
    lds[c * 132 + p1 * 32 + w] = v;
  }
  __syncthreads();
  bf16* dstb = tok + ((size_t)win * 64 + ph * 8) * 1024;
#pragma unroll
  for (int i = 0; i < 32; ++i) {
    int od = i * 256 + tid;                            // pw*1024 + d
    int pw = od >> 10, d = od & 1023;
    int p1 = d >> 8, p2 = (d >> 6) & 3, c = d & 63;
    dstb[(size_t)pw * 1024 + d] = (bf16)lds[c * 132 + p1 * 32 + pw * 4 + p2];
  }
}

// ---------------------------------------------------------------------------
// Kernel 3: fused window attention. One block (8 waves) per window.
// LDS (dynamic, 315392 B):
//   tokc : 2 x [64][136] bf16  (double-buffered token k-chunks)   @ 0
//   Qs   : [64][520] bf16                                         @ 34816
//   Ks   : [64][520] bf16                                         @ 101376
//   Vt   : [8 heads][64 dh][72] bf16 (V transposed)               @ 167936
//   Ps   : [8 waves][64][72] bf16 / overlaid by Os [64][520]      @ 241664
// ---------------------------------------------------------------------------
__global__ __launch_bounds__(256) void win_attn_k(const bf16* __restrict__ tok,
                                                  const bf16* __restrict__ wpack,
                                                  const float* __restrict__ b_out,
                                                  float* __restrict__ out) {
  extern __shared__ unsigned char smem[];
  bf16* tokc = (bf16*)(smem);
  bf16* Qs   = (bf16*)(smem + 34816);
  bf16* Ks   = (bf16*)(smem + 101376);
  bf16* Vt   = (bf16*)(smem + 167936);
  bf16* Ps   = (bf16*)(smem + 241664);
  bf16* Os   = (bf16*)(smem + 241664);

  const int tid = threadIdx.x;
  const int wv = tid >> 5, lane = tid & 31;
  const int l15 = lane & 15, lhi = lane >> 4;          // lhi in {0,1}
  const int win = blockIdx.x;
  const bf16* tokg = tok + (size_t)win * 64 * 1024;
  const v8f z8 = {0.f, 0.f, 0.f, 0.f, 0.f, 0.f, 0.f, 0.f};

  // ================= Stage B: QKV = tokens @ Wqkv (64x1024 @ 1024x1536) ====
  const int mt = wv & 3;                               // wave's M tile (rows)
  const int ntg0 = (wv >> 2) * 16;                     // wave's within-pass N offset
  const int lrow = tid >> 2, lseg = tid & 3;           // cooperative loader map
  const bf16* gsrc = tokg + lrow * 1024 + lseg * 32;   // thread's 64B slice of a chunk row
  bf16* lbase = tokc + lrow * 136 + lseg * 32;

  for (int pass = 0; pass < 3; ++pass) {
    const int ntbase = pass * 32;
    v8f acc[16];
#pragma unroll
    for (int j = 0; j < 16; ++j) acc[j] = z8;

#ifdef HAVE_ASYNC_LDS
    // --- async DMA double-buffered token chunks (no VGPR staging) ---
    {
#pragma unroll
      for (int i = 0; i < 4; ++i)
        __builtin_amdgcn_global_load_async_to_lds_b128(
            AS_GLOBAL(gsrc + i * 8), AS_LOCAL(lbase + i * 8), 0, 0);
      wait_async0();
    }
    __syncthreads();
    for (int c = 0; c < 8; ++c) {
      if (c < 7) {
        const bf16* g = gsrc + (c + 1) * 128;
        bf16* l = lbase + ((c + 1) & 1) * 8704;
#pragma unroll
        for (int i = 0; i < 4; ++i)
          __builtin_amdgcn_global_load_async_to_lds_b128(
              AS_GLOBAL(g + i * 8), AS_LOCAL(l + i * 8), 0, 0);
      }
      const bf16* buf = tokc + (c & 1) * 8704;
#pragma unroll
      for (int kk = 0; kk < 4; ++kk) {
        const int kt = c * 4 + kk;
        v16bf a = ldA(buf + (mt * 16 + l15) * 136 + kk * 32 + lhi * 16);
#pragma unroll
        for (int j = 0; j < 16; ++j) {
          const bf16* bp =
              wpack + ((size_t)(kt * 96 + ntbase + ntg0 + j)) * 512 + lane * 16;
          acc[j] = WMMA_BF16(a, ldB(bp), acc[j]);
        }
      }
      if (c < 7) wait_async0();
      __syncthreads();
    }
#else
    // --- fallback: synchronous copy, no register prefetch (low pressure) ---
    {
      uint4 t0[4];
#pragma unroll
      for (int i = 0; i < 4; ++i) t0[i] = *(const uint4*)(gsrc + i * 8);
#pragma unroll
      for (int i = 0; i < 4; ++i) *(uint4*)(lbase + i * 8) = t0[i];
    }
    __syncthreads();
    for (int c = 0; c < 8; ++c) {
      const bf16* buf = tokc + (c & 1) * 8704;
#pragma unroll
      for (int kk = 0; kk < 4; ++kk) {
        const int kt = c * 4 + kk;
        v16bf a = ldA(buf + (mt * 16 + l15) * 136 + kk * 32 + lhi * 16);
#pragma unroll
        for (int j = 0; j < 16; ++j) {
          const bf16* bp =
              wpack + ((size_t)(kt * 96 + ntbase + ntg0 + j)) * 512 + lane * 16;
          acc[j] = WMMA_BF16(a, ldB(bp), acc[j]);
        }
      }
      __syncthreads();
      if (c < 7) {
        uint4 t1[4];
        const bf16* g = gsrc + (c + 1) * 128;
        bf16* l = lbase + ((c + 1) & 1) * 8704;
#pragma unroll
        for (int i = 0; i < 4; ++i) t1[i] = *(const uint4*)(g + i * 8);
#pragma unroll
        for (int i = 0; i < 4; ++i) *(uint4*)(l + i * 8) = t1[i];
        __syncthreads();
      }
    }
#endif

    // write back this pass's QKV tiles as bf16 into Q / K / V(transposed)
#pragma unroll
    for (int j = 0; j < 16; ++j) {
      const int colb = (ntbase + ntg0 + j) * 16;       // uniform per tile
      const int col = colb + l15;
#pragma unroll
      for (int r = 0; r < 8; ++r) {
        const int m = mt * 16 + r + lhi * 8;
        bf16 v = (bf16)acc[j][r];
        if (colb < 512)        Qs[m * 520 + col] = v;
        else if (colb < 1024)  Ks[m * 520 + (col - 512)] = v;
        else { int vc = col - 1024; Vt[((vc >> 6) * 64 + (vc & 63)) * 72 + m] = v; }
      }
    }
  }
  __syncthreads();

  // ================= Stage C: per-head attention (wave wv = head wv) =======
  {
    const int h = wv;
    v8f sacc[16];
#pragma unroll
    for (int i = 0; i < 16; ++i) sacc[i] = z8;
    // S = Q_h @ K_h^T  (64x64, K-dim = 64)
#pragma unroll
    for (int kk = 0; kk < 2; ++kk) {
#pragma unroll
      for (int mt2 = 0; mt2 < 4; ++mt2) {
        v16bf a = ldA(Qs + (mt2 * 16 + l15) * 520 + h * 64 + kk * 32 + lhi * 16);
#pragma unroll
        for (int nt2 = 0; nt2 < 4; ++nt2) {
          const bf16* bp = Ks + (nt2 * 16 + l15) * 520 + h * 64 + kk * 32 + lhi * 16;
          sacc[mt2 * 4 + nt2] = WMMA_BF16(a, ldB(bp), sacc[mt2 * 4 + nt2]);
        }
      }
    }
    // softmax over rows (row lives in one 16-lane half across 4 n-tiles)
    const float pscale = 0.125f;                        // 1/sqrt(64)
#pragma unroll
    for (int mt2 = 0; mt2 < 4; ++mt2) {
#pragma unroll
      for (int r = 0; r < 8; ++r) {
        float mx = -3.4e38f;
#pragma unroll
        for (int nt2 = 0; nt2 < 4; ++nt2) mx = fmaxf(mx, sacc[mt2 * 4 + nt2][r]);
#pragma unroll
        for (int off = 8; off >= 1; off >>= 1) mx = fmaxf(mx, __shfl_xor(mx, off, 32));
        float sum = 0.f;
#pragma unroll
        for (int nt2 = 0; nt2 < 4; ++nt2) {
          float e = __expf((sacc[mt2 * 4 + nt2][r] - mx) * pscale);
          sacc[mt2 * 4 + nt2][r] = e;
          sum += e;
        }
#pragma unroll
        for (int off = 8; off >= 1; off >>= 1) sum += __shfl_xor(sum, off, 32);
        float inv = 1.f / sum;
#pragma unroll
        for (int nt2 = 0; nt2 < 4; ++nt2) sacc[mt2 * 4 + nt2][r] *= inv;
      }
    }
    // write P (bf16) to this wave's private LDS slice
    bf16* pw = Ps + wv * (64 * 72);
#pragma unroll
    for (int mt2 = 0; mt2 < 4; ++mt2)
#pragma unroll
      for (int nt2 = 0; nt2 < 4; ++nt2)
#pragma unroll
        for (int r = 0; r < 8; ++r)
          pw[(mt2 * 16 + r + lhi * 8) * 72 + nt2 * 16 + l15] =
              (bf16)sacc[mt2 * 4 + nt2][r];

    // O = P @ V_h  (64x64, K-dim = 64); B from transposed V (contiguous runs)
    v8f oacc[16];
#pragma unroll
    for (int i = 0; i < 16; ++i) oacc[i] = z8;
    const bf16* vh = Vt + h * (64 * 72);
#pragma unroll
    for (int kk = 0; kk < 2; ++kk) {
#pragma unroll
      for (int mt2 = 0; mt2 < 4; ++mt2) {
        v16bf a = ldA(pw + (mt2 * 16 + l15) * 72 + kk * 32 + lhi * 16);
#pragma unroll
        for (int nt2 = 0; nt2 < 4; ++nt2) {
          const bf16* bp = vh + (nt2 * 16 + l15) * 72 + kk * 32 + lhi * 16;
          oacc[mt2 * 4 + nt2] = WMMA_BF16(a, ldB(bp), oacc[mt2 * 4 + nt2]);
        }
      }
    }
    __syncthreads();  // all waves done reading Ps before Os overlays it
#pragma unroll
    for (int mt2 = 0; mt2 < 4; ++mt2)
#pragma unroll
      for (int nt2 = 0; nt2 < 4; ++nt2)
#pragma unroll
        for (int r = 0; r < 8; ++r)
          Os[(mt2 * 16 + r + lhi * 8) * 520 + h * 64 + nt2 * 16 + l15] =
              (bf16)oacc[mt2 * 4 + nt2][r];
    __syncthreads();
  }

  // ================= Stage D: out = O @ Wout + b, un-patchify scatter ======
  {
    const int mtd = wv & 3;
    const int ntb = (wv >> 2) * 32;
    const int bidx = win >> 6, h1 = (win >> 3) & 7, w1 = win & 7;
    v16bf afr[16];
#pragma unroll
    for (int kt = 0; kt < 16; ++kt)
      afr[kt] = ldA(Os + (mtd * 16 + l15) * 520 + kt * 32 + lhi * 16);
    for (int j = 0; j < 32; ++j) {
      const int nt = ntb + j;
      v8f acc = z8;
#pragma unroll
      for (int kt = 0; kt < 16; ++kt) {
        const bf16* bp = wpack + (size_t)(3072 + kt * 64 + nt) * 512 + lane * 16;
        acc = WMMA_BF16(afr[kt], ldB(bp), acc);
      }
      const int col = nt * 16 + l15;                   // output dim index
      const float bias = b_out[col];
      const int c = col & 63, p2 = (col >> 6) & 3, p1 = col >> 8;
      const int wbase = w1 * 32 + p2;
#pragma unroll
      for (int r = 0; r < 8; ++r) {
        const int t = mtd * 16 + r + lhi * 8;          // token index
        const int ph = t >> 3, pw2 = t & 7;
        const int himg = h1 * 32 + ph * 4 + p1;
        const int wimg = wbase + pw2 * 4;
        out[(((size_t)(bidx * 64 + c)) * 256 + himg) * 256 + wimg] = acc[r] + bias;
      }
    }
  }
}

// ---------------------------------------------------------------------------
extern "C" void kernel_launch(void* const* d_in, const int* in_sizes, int n_in,
                              void* d_out, int out_size, void* d_ws, size_t ws_size,
                              hipStream_t stream) {
  const float* x    = (const float*)d_in[0];
  const float* wqkv = (const float*)d_in[1];
  const float* wout = (const float*)d_in[2];
  const float* bout = (const float*)d_in[3];
  float* out = (float*)d_out;

  bf16* wpack = (bf16*)d_ws;                             // 4096*512 bf16  (4 MB)
  bf16* tokg  = wpack + (size_t)4096 * 512;              // 512*64*1024 bf16 (64 MB)

  pack_weights_k<<<512, 256, 0, stream>>>(wqkv, wout, wpack);
  gather_tokens_k<<<4096, 256, 0, stream>>>(x, tokg);
  win_attn_k<<<512, 256, 315392, stream>>>(tokg, wpack, bout, out);

  (void)in_sizes; (void)n_in; (void)out_size; (void)ws_size;
}
